// NodeModel_14585708937339
// MI455X (gfx1250) — matrix operand order repaired and checked by reading, not compiled, not run on hardware
//
#include <hip/hip_runtime.h>
#include <cstdint>
#include <cstddef>

typedef __attribute__((ext_vector_type(16))) _Float16 v16h;
typedef __attribute__((ext_vector_type(8)))  _Float16 v8h;
typedef __attribute__((ext_vector_type(8)))  float    v8f;

constexpr int N_NODES = 50000;
constexpr int N_EDGES = 600000;
constexpr int D_NODE  = 128;
constexpr int D_EDGE  = 128;
constexpr int D_IN    = 256;   // D_NODE + D_EDGE
constexpr int D_HID   = 512;
constexpr int D_OUT   = 128;
constexpr int M_PAD   = 50048; // 391 * 128, full 128-row workgroup tiles

// ---- workspace layout (bytes) -------------------------------------------
// sums   : 50000*128*4 = 25,600,000
// cnt    : 50000*4 (slot padded to 262,144)
// h16    : M_PAD*256*2 = 25,624,576
// hB     : M_PAD*512*2 = 51,249,152  (aliases sums+cnt+h16, all dead by GEMM2)
// hA     : M_PAD*512*2 = 51,249,152
// W1t/W2t/W3t f16 transposed
constexpr size_t OFF_SUMS = 0;
constexpr size_t OFF_CNT  = 25600000;
constexpr size_t OFF_H16  = 25862144;
constexpr size_t OFF_HB   = 0;
constexpr size_t OFF_HA   = 51486720;
constexpr size_t OFF_W1T  = 102735872;                 // 256*512*2 = 262,144
constexpr size_t OFF_W2T  = OFF_W1T + 262144;          // 512*512*2 = 524,288
constexpr size_t OFF_W3T  = OFF_W2T + 524288;          // 512*128*2 = 131,072
// total ~103.7 MB

// -------------------------------------------------------------------------
__global__ void zero_f32(float* __restrict__ p, int n) {
    int i = blockIdx.x * blockDim.x + threadIdx.x;
    if (i < n) p[i] = 0.0f;
}

// scatter-add edge_attr into sums[recv], count edges per receiver.
// 32 threads per edge, 4 floats per thread.
__global__ void scatter_add(const float* __restrict__ ea,
                            const long long* __restrict__ recv,
                            float* __restrict__ sums, float* __restrict__ cnt) {
    int tid = blockIdx.x * blockDim.x + threadIdx.x;
    int e   = tid >> 5;
    if (e >= N_EDGES) return;
    int c4  = (tid & 31) << 2;
    int r   = (int)recv[e];
    const float4 v = *(const float4*)(ea + (size_t)e * D_EDGE + c4);
    float* dst = sums + (size_t)r * D_EDGE + c4;
    atomicAdd(dst + 0, v.x);
    atomicAdd(dst + 1, v.y);
    atomicAdd(dst + 2, v.z);
    atomicAdd(dst + 3, v.w);
    if ((tid & 31) == 0) atomicAdd(cnt + r, 1.0f);
}

// f32 [K][N] row-major -> f16 transposed [N][K] (so B fragments are contiguous in K)
__global__ void w_to_f16T(const float* __restrict__ W, _Float16* __restrict__ Wt,
                          int K, int N) {
    int tid = blockIdx.x * blockDim.x + threadIdx.x;
    if (tid >= K * N) return;
    int k = tid / N, n = tid % N;
    Wt[(size_t)n * K + k] = (_Float16)W[(size_t)k * N + n];
}

// h16[m][0:128) = x[m], h16[m][128:256) = sums[m]/max(cnt,1); padded rows -> 0
__global__ void build_h16(const float* __restrict__ x, const float* __restrict__ sums,
                          const float* __restrict__ cnt, _Float16* __restrict__ h16) {
    int tid = blockIdx.x * blockDim.x + threadIdx.x;   // M_PAD*64 threads
    int m   = tid >> 6;
    if (m >= M_PAD) return;
    int c4 = (tid & 63) << 2;
    float4 v = make_float4(0.f, 0.f, 0.f, 0.f);
    if (m < N_NODES) {
        if (c4 < D_NODE) {
            v = *(const float4*)(x + (size_t)m * D_NODE + c4);
        } else {
            float s = 1.0f / fmaxf(cnt[m], 1.0f);
            float4 t = *(const float4*)(sums + (size_t)m * D_EDGE + (c4 - D_NODE));
            v = make_float4(t.x * s, t.y * s, t.z * s, t.w * s);
        }
    }
    _Float16* dst = h16 + (size_t)m * D_IN + c4;
    dst[0] = (_Float16)v.x; dst[1] = (_Float16)v.y;
    dst[2] = (_Float16)v.z; dst[3] = (_Float16)v.w;
}

// -------------------------------------------------------------------------
// WMMA GEMM: out[M_PAD x N] = relu?(A[M_PAD x K] @ Bt^T + bias)
// Bt is f16 [N][K]. 8 waves/block; each wave computes a 32(M) x 64(N) tile
// with 2x4 v_wmma_f32_16x16x32_f16 accumulators. Fragment addressing follows
// the CDNA5 ISA 16-bit A/B/C layouts (CDNA5 ISA §7.12.2).
template<bool RELU, bool OUT32>
__global__ __launch_bounds__(256)
void gemm_wmma(const _Float16* __restrict__ A, const _Float16* __restrict__ Bt,
               const float* __restrict__ bias, void* __restrict__ outv,
               int K, int N, int Mreal) {
    const int lane = threadIdx.x & 31;
    const int wave = threadIdx.x >> 5;   // 0..7
    const int wm   = wave >> 1;          // 0..3
    const int wn   = wave & 1;           // 0..1
    const int l15  = lane & 15;
    const int hi16 = lane >> 4;          // 0 or 1

    const int mBase = blockIdx.x * 128 + wm * 32;
    const int nBase = blockIdx.y * 128 + wn * 64;

    v8f acc[2][4];
    for (int i = 0; i < 2; ++i)
        for (int j = 0; j < 4; ++j)
            for (int r = 0; r < 8; ++r) acc[i][j][r] = 0.0f;

    const _Float16* Arow0 = A + (size_t)(mBase + l15) * K;
    const _Float16* Arow1 = A + (size_t)(mBase + 16 + l15) * K;
    const _Float16* Brow[4];
#pragma unroll
    for (int j = 0; j < 4; ++j) Brow[j] = Bt + (size_t)(nBase + j * 16 + l15) * K;

    for (int k0 = 0; k0 < K; k0 += 32) {
        // A fragment (16-bit A 16x32): lanes<16 hold K {0..7,16..23}, lanes>=16 {8..15,24..31}
        const int kbA = k0 + hi16 * 8;
        // B fragment (32x16): lanes<16 hold K {0..7,16..23}, lanes>=16 {8..15,24..31}, N = lane&15
        const int kbB = k0 + hi16 * 8;

        union { v16h v; v8h h[2]; } a[2], b[4];
        a[0].h[0] = *(const v8h*)(Arow0 + kbA);
        a[0].h[1] = *(const v8h*)(Arow0 + kbA + 16);
        a[1].h[0] = *(const v8h*)(Arow1 + kbA);
        a[1].h[1] = *(const v8h*)(Arow1 + kbA + 16);
#pragma unroll
        for (int j = 0; j < 4; ++j) {
            b[j].h[0] = *(const v8h*)(Brow[j] + kbB);
            b[j].h[1] = *(const v8h*)(Brow[j] + kbB + 16);
        }
        __builtin_prefetch(Arow0 + kbA + 32, 0, 3);  // global_prefetch_b8 next K-slab

#pragma unroll
        for (int i = 0; i < 2; ++i)
#pragma unroll
            for (int j = 0; j < 4; ++j)
                acc[i][j] = __builtin_amdgcn_wmma_f32_16x16x32_f16(
                    false, a[i].v, false, b[j].v, (short)0, acc[i][j], false, false);
    }

    // Epilogue. C/D layout: VGPR r -> M = r (lanes 0-15) or r+8 (lanes 16-31), N = lane&15.
#pragma unroll
    for (int i = 0; i < 2; ++i) {
#pragma unroll
        for (int j = 0; j < 4; ++j) {
            const int col = nBase + j * 16 + l15;
            const float bj = bias[col];
#pragma unroll
            for (int r = 0; r < 8; ++r) {
                const int row = mBase + i * 16 + hi16 * 8 + r;
                float v = acc[i][j][r] + bj;
                if (RELU) v = v > 0.0f ? v : 0.0f;
                if (OUT32) {
                    if (row < Mreal) ((float*)outv)[(size_t)row * N + col] = v;
                } else {
                    ((_Float16*)outv)[(size_t)row * N + col] = (_Float16)v;
                }
            }
        }
    }
}

// -------------------------------------------------------------------------
extern "C" void kernel_launch(void* const* d_in, const int* in_sizes, int n_in,
                              void* d_out, int out_size, void* d_ws, size_t ws_size,
                              hipStream_t stream) {
    const float*     x  = (const float*)d_in[0];
    const long long* ei = (const long long*)d_in[1];   // int64 [2, E]
    const float*     ea = (const float*)d_in[2];
    const float*     W1 = (const float*)d_in[3];
    const float*     b1 = (const float*)d_in[4];
    const float*     W2 = (const float*)d_in[5];
    const float*     b2 = (const float*)d_in[6];
    const float*     W3 = (const float*)d_in[7];
    const float*     b3 = (const float*)d_in[8];
    float*           out = (float*)d_out;

    char* ws = (char*)d_ws;
    float*    sums = (float*)(ws + OFF_SUMS);
    float*    cnt  = (float*)(ws + OFF_CNT);
    _Float16* h16  = (_Float16*)(ws + OFF_H16);
    _Float16* hA   = (_Float16*)(ws + OFF_HA);
    _Float16* hB   = (_Float16*)(ws + OFF_HB);
    _Float16* W1t  = (_Float16*)(ws + OFF_W1T);
    _Float16* W2t  = (_Float16*)(ws + OFF_W2T);
    _Float16* W3t  = (_Float16*)(ws + OFF_W3T);

    // zero sums + cnt region
    const int nz = (int)(OFF_H16 / 4);
    zero_f32<<<(nz + 255) / 256, 256, 0, stream>>>((float*)ws, nz);

    // weights -> f16 transposed [N][K]
    w_to_f16T<<<(D_IN  * D_HID + 255) / 256, 256, 0, stream>>>(W1, W1t, D_IN,  D_HID);
    w_to_f16T<<<(D_HID * D_HID + 255) / 256, 256, 0, stream>>>(W2, W2t, D_HID, D_HID);
    w_to_f16T<<<(D_HID * D_OUT + 255) / 256, 256, 0, stream>>>(W3, W3t, D_HID, D_OUT);

    // scatter-mean numerator/denominator
    scatter_add<<<(N_EDGES * 32) / 256, 256, 0, stream>>>(ea, ei + N_EDGES, sums, cnt);

    // concat [x | mean] in f16, padded rows zeroed
    build_h16<<<(M_PAD * 64) / 256, 256, 0, stream>>>(x, sums, cnt, h16);

    // MLP via WMMA
    dim3 blk(256);
    gemm_wmma<true,  false><<<dim3(M_PAD / 128, D_HID / 128), blk, 0, stream>>>(
        h16, W1t, b1, (void*)hA, D_IN,  D_HID, M_PAD);
    gemm_wmma<true,  false><<<dim3(M_PAD / 128, D_HID / 128), blk, 0, stream>>>(
        hA,  W2t, b2, (void*)hB, D_HID, D_HID, M_PAD);
    gemm_wmma<false, true ><<<dim3(M_PAD / 128, D_OUT / 128), blk, 0, stream>>>(
        hB,  W3t, b3, (void*)out, D_HID, D_OUT, N_NODES);
}
